// AttentionBlock_1649267442223
// MI455X (gfx1250) — compile-verified, matrix operand order checked by reading, not run on hardware
//
#include <hip/hip_runtime.h>
#include <hip/hip_bf16.h>
#include <math.h>

// ---------------------------------------------------------------------------
// CDNA5 (gfx1250) WMMA attention block.
// D = A(16x32 f16) x B(32x16 f16) + C(16x16 f32), wave32.
// ---------------------------------------------------------------------------

typedef _Float16 v4h  __attribute__((ext_vector_type(4)));
typedef _Float16 v8h  __attribute__((ext_vector_type(8)));
typedef _Float16 v16h __attribute__((ext_vector_type(16)));
typedef float    v4f  __attribute__((ext_vector_type(4)));
typedef float    v8f  __attribute__((ext_vector_type(8)));

__device__ __forceinline__ v16h cat16(v8h lo, v8h hi) {
  return __builtin_shufflevector(lo, hi, 0,1,2,3,4,5,6,7,8,9,10,11,12,13,14,15);
}
__device__ __forceinline__ v8h ldv8(const _Float16* p) {
  return *(const v8h*)p;
}
__device__ __forceinline__ v8f wmma16(v16h a, v16h b, v8f c) {
  return __builtin_amdgcn_wmma_f32_16x16x32_f16(false, a, false, b, (short)0, c, false, false);
}

// A/B-style 16x32 fragment from an LDS tile: lane L -> row (L%16), K pairs at
// half*8 and 16+half*8. rowStride must be a multiple of 8 halves (16B) so the
// two reads lower to aligned ds_load_b128.
__device__ __forceinline__ v16h frag32(const _Float16* lds, int row, int rowStride,
                                       int koff, int half) {
  const _Float16* p = lds + row * rowStride + koff + half * 8;
  return cat16(ldv8(p), ldv8(p + 16));
}

// ---------------------------------------------------------------------------
// Kernel 1: GroupNorm (B=4, G=8, C/G=32, N=4096) -> f16 normalized output
// ---------------------------------------------------------------------------
__global__ void gn_kernel(const float* __restrict__ x,
                          const float* __restrict__ gamma,
                          const float* __restrict__ beta,
                          _Float16* __restrict__ h,
                          int C, int N, int Cg) {
  const int G  = C / Cg;
  const int bg = blockIdx.x;
  const int b  = bg / G;
  const int g  = bg % G;
  const long base  = ((long)b * C + (long)g * Cg) * N;
  const int  total = Cg * N;
  const int  nv    = total >> 2;   // float4 elements

  float s = 0.f, ss = 0.f;
  for (int i = threadIdx.x; i < nv; i += blockDim.x) {
    v4f v = *(const v4f*)&x[base + i * 4];
#pragma unroll
    for (int u = 0; u < 4; ++u) { s += v[u]; ss += v[u] * v[u]; }
  }
  __shared__ float rs[256];
  __shared__ float rq[256];
  rs[threadIdx.x] = s;
  rq[threadIdx.x] = ss;
  __syncthreads();
  for (int off = 128; off > 0; off >>= 1) {
    if ((int)threadIdx.x < off) {
      rs[threadIdx.x] += rs[threadIdx.x + off];
      rq[threadIdx.x] += rq[threadIdx.x + off];
    }
    __syncthreads();
  }
  const float mean = rs[0] / (float)total;
  const float var  = rq[0] / (float)total - mean * mean;
  const float inv  = rsqrtf(var + 1e-5f);

  for (int i = threadIdx.x; i < nv; i += blockDim.x) {
    int c = g * Cg + (i * 4) / N;   // N % 4 == 0: channel constant over the 4
    float ga = gamma[c], be = beta[c];
    v4f v = *(const v4f*)&x[base + i * 4];
    v4h o;
#pragma unroll
    for (int u = 0; u < 4; ++u) o[u] = (_Float16)((v[u] - mean) * inv * ga + be);
    *(v4h*)&h[base + i * 4] = o;
  }
}

// ---------------------------------------------------------------------------
// Kernel 2/4: WMMA GEMM  out[b] = W[MxK] * Bm[b][KxN] + bias (+ resid)
// Block tile 64x64, 8 waves (2 M-waves x 4 N-waves), K-step 32.
// ---------------------------------------------------------------------------
template <typename OutT, bool RES>
__global__ void gemm_kernel(const float* __restrict__ Wm,
                            const _Float16* __restrict__ Bm,
                            const float* __restrict__ bias,
                            const float* __restrict__ resid,
                            OutT* __restrict__ out,
                            int Kd, int Nn,
                            long strideB, long strideO, long strideR) {
  __shared__ alignas(16) _Float16 Al[64 * 48];   // A tile [m][k], stride 48 (96B)
  __shared__ alignas(16) _Float16 Bt[64 * 48];   // B tile transposed [n][k]
  __shared__ float BiasL[64];

  const int t    = threadIdx.x;
  const int w    = t >> 5;
  const int ln   = t & 31;
  const int lm   = ln & 15;
  const int half = ln >> 4;
  const int mw   = w & 1;   // M sub-block of 32
  const int nw   = w >> 1;  // N tile of 16

  const int  b  = blockIdx.z;
  const int  m0 = blockIdx.x * 64;
  const int  n0 = blockIdx.y * 64;
  const _Float16* Bp = Bm + (long)b * strideB;
  const float*    Rp = RES ? (resid + (long)b * strideR) : nullptr;
  OutT*           Op = out + (long)b * strideO;

  if (t < 64) BiasL[t] = bias[m0 + t];

  v8f c0 = {};
  v8f c1 = {};

  for (int k0 = 0; k0 < Kd; k0 += 32) {
    __syncthreads();
    // Stage A (f32 -> f16): 64x32, float4 global loads, b64 LDS stores.
#pragma unroll
    for (int j = 0; j < 2; ++j) {
      int idx = t + j * 256;           // 0..511
      int r   = idx >> 3;              // 0..63
      int cc  = (idx & 7) * 4;         // 0..28
      v4f w4 = *(const v4f*)&Wm[(long)(m0 + r) * Kd + k0 + cc];
      v4h h4;
#pragma unroll
      for (int u = 0; u < 4; ++u) h4[u] = (_Float16)w4[u];
      *(v4h*)&Al[r * 48 + cc] = h4;
    }
    // Stage B transposed: 32x64 -> Bt[n][k], v4h global loads.
#pragma unroll
    for (int j = 0; j < 2; ++j) {
      int idx = t + j * 256;           // 0..511
      int kk  = idx >> 4;              // 0..31
      int nn  = (idx & 15) * 4;        // 0..60
      v4h b4 = *(const v4h*)&Bp[(long)(k0 + kk) * Nn + n0 + nn];
#pragma unroll
      for (int u = 0; u < 4; ++u) Bt[(nn + u) * 48 + kk] = b4[u];
    }
    __syncthreads();

    const int ma = mw * 32 + lm;
    v16h a0 = frag32(Al, ma,      48, 0, half);
    v16h a1 = frag32(Al, ma + 16, 48, 0, half);
    v16h bb = frag32(Bt, nw * 16 + lm, 48, 0, half);
    c0 = wmma16(a0, bb, c0);
    c1 = wmma16(a1, bb, c1);
  }

  // Epilogue
  const int gcol = n0 + nw * 16 + lm;
#pragma unroll
  for (int r = 0; r < 8; ++r) {
    int row0 = mw * 32 + r + 8 * half;
    int row1 = row0 + 16;
    float v0 = c0[r] + BiasL[row0];
    float v1 = c1[r] + BiasL[row1];
    if (RES) {
      v0 += Rp[(long)(m0 + row0) * Nn + gcol];
      v1 += Rp[(long)(m0 + row1) * Nn + gcol];
    }
    Op[(long)(m0 + row0) * Nn + gcol] = (OutT)v0;
    Op[(long)(m0 + row1) * Nn + gcol] = (OutT)v1;
  }
}

// ---------------------------------------------------------------------------
// Kernel 3: flash attention. One block = 128 queries of one (b, head).
// Each wave owns 16 queries end-to-end (softmax rows live in one half-wave).
// ---------------------------------------------------------------------------
__global__ void attn_kernel(const _Float16* __restrict__ qkv,
                            _Float16* __restrict__ o,
                            int C, int N, int hd) {
  __shared__ alignas(16) _Float16 Qt[128 * 80];     // [query][d], stride 80 (160B)
  __shared__ alignas(16) _Float16 Kt[64 * 80];      // [key][d]
  __shared__ alignas(16) _Float16 Vl[64 * 64];      // [d][key], stride 64 (128B)
  __shared__ alignas(16) _Float16 Psc[8 * 16 * 64]; // per-wave P scratch

  const int t    = threadIdx.x;
  const int w    = t >> 5;
  const int ln   = t & 31;
  const int lm   = ln & 15;
  const int half = ln >> 4;

  const int qb = blockIdx.x * 128;
  const int hh = blockIdx.y;
  const int b  = blockIdx.z;
  const float scale = rsqrtf((float)hd);

  const long qbase = ((long)b * 3 * C +         hh * hd) * N;
  const long kbase = ((long)b * 3 * C + C +     hh * hd) * N;
  const long vbase = ((long)b * 3 * C + 2 * C + hh * hd) * N;

  // Stage Q tile transposed to [query][d]: v8h (16B) global loads.
  for (int idx = t; idx < 1024; idx += 256) {
    int d  = idx >> 4;                 // 0..63
    int qi = (idx & 15) * 8;           // 0..120
    v8h q8 = *(const v8h*)&qkv[qbase + (long)d * N + qb + qi];
#pragma unroll
    for (int u = 0; u < 8; ++u) Qt[(qi + u) * 80 + d] = q8[u];
  }
  __syncthreads();

  v8f oacc[4] = {};
  float mrow[8], lrow[8];
#pragma unroll
  for (int r = 0; r < 8; ++r) { mrow[r] = -3.0e38f; lrow[r] = 0.f; }

  _Float16* Pw = &Psc[w * 16 * 64];
  const int qrow = w * 16 + lm;

  for (int m0 = 0; m0 < N; m0 += 64) {
    // Stage K (transposed) and V (native): v8h global loads;
    // V row chunks stay contiguous -> single ds_store_b128.
#pragma unroll
    for (int j = 0; j < 2; ++j) {
      int idx = t + j * 256;           // 0..511
      int d   = idx >> 3;              // 0..63
      int mm  = (idx & 7) * 8;         // 0..56
      v8h k8 = *(const v8h*)&qkv[kbase + (long)d * N + m0 + mm];
      v8h v8 = *(const v8h*)&qkv[vbase + (long)d * N + m0 + mm];
#pragma unroll
      for (int u = 0; u < 8; ++u) Kt[(mm + u) * 80 + d] = k8[u];
      *(v8h*)&Vl[d * 64 + mm] = v8;
    }
    // Prefetch next tile: one 128B line covers a full 64-key f16 row.
    if (m0 + 64 < N && t < 128) {
      int d = t & 63;
      const _Float16* pf = (t < 64) ? &qkv[kbase + (long)d * N + m0 + 64]
                                    : &qkv[vbase + (long)d * N + m0 + 64];
      __builtin_prefetch(pf, 0, 1);
    }
    __syncthreads();

    // S = Q^T K  (16 queries x 64 keys), head_dim 64 = 2 K-steps of 32
    v8f s[4];
    v16h aq0 = frag32(Qt, qrow, 80,  0, half);
    v16h aq1 = frag32(Qt, qrow, 80, 32, half);
#pragma unroll
    for (int tt = 0; tt < 4; ++tt) {
      int krow = tt * 16 + lm;
      v16h b0 = frag32(Kt, krow, 80,  0, half);
      v16h b1 = frag32(Kt, krow, 80, 32, half);
      v8f acc = {};
      acc = wmma16(aq0, b0, acc);
      acc = wmma16(aq1, b1, acc);
      s[tt] = acc;
    }

    // Online softmax: row M=r+8*half of the 16x64 score block lives in VGPR
    // slot r across the 16 lanes of this half-wave.
#pragma unroll
    for (int r = 0; r < 8; ++r) {
      float v = fmaxf(fmaxf(s[0][r], s[1][r]), fmaxf(s[2][r], s[3][r]));
#pragma unroll
      for (int msk = 8; msk >= 1; msk >>= 1)
        v = fmaxf(v, __shfl_xor(v, msk, 32));
      float mnew = fmaxf(mrow[r], v * scale);
      float f    = __expf(mrow[r] - mnew);
      mrow[r]    = mnew;
      float rsum = 0.f;
#pragma unroll
      for (int tt = 0; tt < 4; ++tt) {
        float p = __expf(s[tt][r] * scale - mnew);
        s[tt][r] = p;
        rsum += p;
      }
#pragma unroll
      for (int msk = 8; msk >= 1; msk >>= 1)
        rsum += __shfl_xor(rsum, msk, 32);
      lrow[r] = lrow[r] * f + rsum;
      oacc[0][r] *= f; oacc[1][r] *= f; oacc[2][r] *= f; oacc[3][r] *= f;
    }

    // C-layout f32 -> A-layout f16 via per-wave LDS scratch
#pragma unroll
    for (int tt = 0; tt < 4; ++tt)
#pragma unroll
      for (int r = 0; r < 8; ++r)
        Pw[(r + 8 * half) * 64 + tt * 16 + lm] = (_Float16)s[tt][r];

    // O += P @ V^T   (16 queries x 64 d), 64 keys = 2 K-steps of 32
    v16h ap0 = frag32(Pw, lm, 64,  0, half);
    v16h ap1 = frag32(Pw, lm, 64, 32, half);
#pragma unroll
    for (int tt = 0; tt < 4; ++tt) {
      int vrow = tt * 16 + lm;
      v16h b0 = frag32(Vl, vrow, 64,  0, half);
      v16h b1 = frag32(Vl, vrow, 64, 32, half);
      v8f acc = oacc[tt];
      acc = wmma16(ap0, b0, acc);
      acc = wmma16(ap1, b1, acc);
      oacc[tt] = acc;
    }
    __syncthreads();
  }

  // Normalize + store. Per lane, r=0..7 are contiguous queries at fixed d:
  // pack into v8h and emit one global_store_b128 per O tile.
#pragma unroll
  for (int r = 0; r < 8; ++r) {
    float inv = 1.0f / lrow[r];
    oacc[0][r] *= inv; oacc[1][r] *= inv; oacc[2][r] *= inv; oacc[3][r] *= inv;
  }
  const long obase = ((long)b * C + hh * hd) * N;
  const int  q0    = qb + w * 16 + 8 * half;
#pragma unroll
  for (int tt = 0; tt < 4; ++tt) {
    int d = tt * 16 + lm;
    v8h pack;
#pragma unroll
    for (int r = 0; r < 8; ++r) pack[r] = (_Float16)oacc[tt][r];
    *(v8h*)&o[obase + (long)d * N + q0] = pack;
  }
}

// ---------------------------------------------------------------------------
// Launch
// ---------------------------------------------------------------------------
extern "C" void kernel_launch(void* const* d_in, const int* in_sizes, int n_in,
                              void* d_out, int out_size, void* d_ws, size_t ws_size,
                              hipStream_t stream) {
  const float* x      = (const float*)d_in[0];
  const float* gamma  = (const float*)d_in[1];
  const float* beta   = (const float*)d_in[2];
  const float* qkv_w  = (const float*)d_in[3];
  const float* qkv_b  = (const float*)d_in[4];
  const float* proj_w = (const float*)d_in[5];
  const float* proj_b = (const float*)d_in[6];
  float* out = (float*)d_out;

  const int B = 4, C = 256, N = 64 * 64, G = 8, heads = 4, hd = 64;

  char* ws = (char*)d_ws;
  _Float16* hbuf   = (_Float16*)ws;                                  // B*C*N
  _Float16* qkvbuf = (_Float16*)(ws + (size_t)B * C * N * 2);        // B*3C*N
  _Float16* obuf   = (_Float16*)(ws + (size_t)B * C * N * 2 * 4);    // B*C*N
  (void)in_sizes; (void)n_in; (void)out_size; (void)ws_size;

  // 1) GroupNorm -> f16
  gn_kernel<<<B * G, 256, 0, stream>>>(x, gamma, beta, hbuf, C, N, C / G);

  // 2) QKV = qkv_w [768x256] * h [256x4096] + qkv_b  -> f16
  gemm_kernel<_Float16, false><<<dim3((3 * C) / 64, N / 64, B), 256, 0, stream>>>(
      qkv_w, hbuf, qkv_b, nullptr, qkvbuf,
      C, N, (long)C * N, (long)3 * C * N, 0L);

  // 3) Flash attention per (b, head), 128 queries per block
  attn_kernel<<<dim3(N / 128, heads, B), 256, 0, stream>>>(qkvbuf, obuf, C, N, hd);

  // 4) out = x + proj_w [256x256] * o [256x4096] + proj_b  -> f32
  gemm_kernel<float, true><<<dim3(C / 64, N / 64, B), 256, 0, stream>>>(
      proj_w, obuf, proj_b, x, out,
      C, N, (long)C * N, (long)C * N, (long)C * N);
}
